// SelfAttention_24902220382827
// MI455X (gfx1250) — matrix operand order repaired and checked by reading, not compiled, not run on hardware
//
#include <hip/hip_runtime.h>

// ---------------------------------------------------------------------------
// Self-attention (L2-normed q/k, learned per-head scale) for MI455X / gfx1250.
// All matmuls on v_wmma_f32_16x16x32_bf16 (f32 accumulate), flash-attention
// (no [B,H,L,L] materialization). LDS tiles are double-buffered and filled
// with GLOBAL_LOAD_ASYNC_TO_LDS_B128 (ASYNCcnt) so copies overlap WMMA work.
// wave32 throughout.
// ---------------------------------------------------------------------------

#define BB 2
#define LL 2048
#define CC 1024
#define HH 16
#define DD 64
#define MAX_SCALE_MUL 4.605170185988092f   // log(100)

typedef __bf16 bf16;
typedef __attribute__((ext_vector_type(8)))  __bf16 bf16x8;
typedef __attribute__((ext_vector_type(16))) __bf16 bf16x16;
typedef __attribute__((ext_vector_type(8)))  float  floatx8;

__device__ __forceinline__ bf16x16 join8(bf16x8 lo, bf16x8 hi) {
  bf16x16 r;
#pragma unroll
  for (int i = 0; i < 8; ++i) { r[i] = lo[i]; r[i + 8] = hi[i]; }
  return r;
}

// A-matrix fragment (16x32 bf16, K along k0..k0+31) from a row-major buffer.
// Layout: lanes 0-15 row M=lane hold K {0..15}; lanes 16-31 row
// M=lane-16 hold K {16..31}.
__device__ __forceinline__ bf16x16 load_afrag(const bf16* __restrict__ base,
                                              int stride, int row, int k0, int half) {
  const bf16* p = base + (size_t)row * stride + k0 + half * 16;
  bf16x8 lo = *reinterpret_cast<const bf16x8*>(p);
  bf16x8 hi = *reinterpret_cast<const bf16x8*>(p + 8);
  return join8(lo, hi);
}

// B-matrix fragment (32x16 bf16): column n = lane%16, elements i = K(half*16+i)
// -> one contiguous 16-element (32B) read per lane.
__device__ __forceinline__ bf16x16 load_bfrag(const bf16* p) {
  bf16x8 lo = *reinterpret_cast<const bf16x8*>(p);
  bf16x8 hi = *reinterpret_cast<const bf16x8*>(p + 8);
  return join8(lo, hi);
}

__device__ __forceinline__ floatx8 wmma_bf16(bf16x16 a, bf16x16 b, floatx8 c) {
  return __builtin_amdgcn_wmma_f32_16x16x32_bf16(false, a, false, b,
                                                 (short)0, c, false, false);
}

// CDNA5 async global->LDS copy, 16B per lane (GLOBAL_LOAD_ASYNC_TO_LDS_B128,
// tracked by ASYNCcnt). LDS address = low 32 bits of the generic pointer.
__device__ __forceinline__ void async_copy_b128(void* lds_dst, const void* gsrc) {
  asm volatile("global_load_async_to_lds_b128 %0, %1, off"
               :: "v"((unsigned int)(uintptr_t)lds_dst),
                  "v"((unsigned long long)(uintptr_t)gsrc)
               : "memory");
}
__device__ __forceinline__ void wait_async0() {
  asm volatile("s_wait_asynccnt 0x0" ::: "memory");
}

__device__ __forceinline__ float redsum16(float v) {
#pragma unroll
  for (int m = 1; m < 16; m <<= 1) v += __shfl_xor(v, m, 16);
  return v;
}
__device__ __forceinline__ float redmax16(float v) {
#pragma unroll
  for (int m = 1; m < 16; m <<= 1) v = fmaxf(v, __shfl_xor(v, m, 16));
  return v;
}

// ---------------------------------------------------------------------------
// fp32 -> bf16 conversion (vectorized x4)
// ---------------------------------------------------------------------------
__global__ void cvt_bf16_kernel(const float* __restrict__ in,
                                bf16* __restrict__ out, int n4) {
  int i = blockIdx.x * blockDim.x + threadIdx.x;
  if (i < n4) {
    float4 f = reinterpret_cast<const float4*>(in)[i];
    bf16 o0 = (bf16)f.x, o1 = (bf16)f.y, o2 = (bf16)f.z, o3 = (bf16)f.w;
    bf16* p = out + (size_t)i * 4;
    p[0] = o0; p[1] = o1; p[2] = o2; p[3] = o3;
  }
}

// ---------------------------------------------------------------------------
// QKV GEMM: qkv = x @ w_qkv^T + bias, fused per-head L2-norm (+ exp(scale))
// for q/k. Block = 4 waves, each wave 16 rows x 64 cols (one head slice).
// W tiles double-buffered in LDS via async copy; A fragments prefetched one
// K-step ahead.
// ---------------------------------------------------------------------------
__global__ __launch_bounds__(128) void qkv_gemm_kernel(
    const bf16* __restrict__ xb, const bf16* __restrict__ wb,
    const float* __restrict__ q_bias, const float* __restrict__ v_bias,
    const float* __restrict__ scale_mul,
    bf16* __restrict__ Qb, bf16* __restrict__ Kb, bf16* __restrict__ Vb) {
  __shared__ bf16 w_lds[2][64 * 32];
  const int lane = threadIdx.x & 31, wave = threadIdx.x >> 5;
  const int ln = lane & 15, half = lane >> 4;
  const int row0 = blockIdx.x * 64 + wave * 16;     // row into [B*L, C]
  const int n0 = blockIdx.y * 64;                   // col into [3C]

  auto stage_w = [&](int buf, int kk2) {
#pragma unroll
    for (int i = 0; i < 2; ++i) {
      int idx = threadIdx.x + i * 128;              // 256 x 8-elem chunks
      int n = idx >> 2, c8 = (idx & 3) * 8;
      async_copy_b128(&w_lds[buf][n * 32 + c8],
                      &wb[(size_t)(n0 + n) * CC + kk2 + c8]);
    }
  };

  stage_w(0, 0);
  bf16x16 afrag = load_afrag(xb, CC, row0 + ln, 0, half);

  floatx8 acc[4] = {};
  for (int kk = 0; kk < CC; kk += 32) {
    const int cur = (kk >> 5) & 1;
    wait_async0();
    __syncthreads();
    bf16x16 afrag_next = afrag;
    if (kk + 32 < CC) {
      stage_w(cur ^ 1, kk + 32);                    // overlaps with WMMAs below
      afrag_next = load_afrag(xb, CC, row0 + ln, kk + 32, half);
    }
#pragma unroll
    for (int t = 0; t < 4; ++t) {
      bf16x16 bfrag = load_bfrag(&w_lds[cur][(t * 16 + ln) * 32 + half * 16]);
      acc[t] = wmma_bf16(afrag, bfrag, acc[t]);
    }
    afrag = afrag_next;
  }

  const int sec = n0 / CC;        // 0=q 1=k 2=v
  const int h = (n0 % CC) / 64;
  // bias (reference: concat[q_bias, 0, v_bias]) added BEFORE the norm
  float badd[4];
#pragma unroll
  for (int t = 0; t < 4; ++t) {
    int d = t * 16 + ln;
    badd[t] = (sec == 0) ? q_bias[h * 64 + d] : (sec == 2 ? v_bias[h * 64 + d] : 0.f);
  }
#pragma unroll
  for (int t = 0; t < 4; ++t)
#pragma unroll
    for (int r = 0; r < 8; ++r) acc[t][r] += badd[t];

  if (sec < 2) {  // L2-normalize each row over D=64 (exactly this wave's tile)
    float sm = 1.f;
    if (sec == 0) sm = __expf(fminf(scale_mul[h], MAX_SCALE_MUL));
#pragma unroll
    for (int r = 0; r < 8; ++r) {
      float sq = 0.f;
#pragma unroll
      for (int t = 0; t < 4; ++t) sq += acc[t][r] * acc[t][r];
      sq = redsum16(sq);
      float inv = sm / fmaxf(sqrtf(sq), 1e-12f);
#pragma unroll
      for (int t = 0; t < 4; ++t) acc[t][r] *= inv;
    }
  }

  bf16* outb = (sec == 0) ? Qb : (sec == 1 ? Kb : Vb);
#pragma unroll
  for (int r = 0; r < 8; ++r) {
    int rg = row0 + r + 8 * half;
    int b = rg >> 11;                 // / L
    int l = rg & (LL - 1);
    size_t base = (((size_t)b * HH + h) * LL + l) * DD;
#pragma unroll
    for (int t = 0; t < 4; ++t) outb[base + t * 16 + ln] = (bf16)acc[t][r];
  }
}

// ---------------------------------------------------------------------------
// Flash attention: block = 4 waves on one (b,h), 64 consecutive q rows
// (16 per wave). Per 32-key step: K tile async-copied to LDS (double
// buffered), V tile prefetched to VGPRs and stored transposed after the
// compute, S = QK^T (4 WMMA), bias + online softmax, P via per-wave LDS tile
// (C-layout -> A-layout), O += P V (4 WMMA).
// ---------------------------------------------------------------------------
__global__ __launch_bounds__(128) void attn_kernel(
    const bf16* __restrict__ Qb, const bf16* __restrict__ Kb,
    const bf16* __restrict__ Vb, const float* __restrict__ attn_bias,
    bf16* __restrict__ Ob) {
  __shared__ bf16 k_lds[2][32 * 64];     // [m][d]
  __shared__ bf16 vT_lds[2][64 * 32];    // [d][m]
  __shared__ bf16 p_lds[4][16 * 32];     // per-wave P tile [row][m_local]

  const int lane = threadIdx.x & 31, wave = threadIdx.x >> 5;
  const int ln = lane & 15, half = lane >> 4;
  const int h = blockIdx.y, b = blockIdx.z;
  const int l0 = blockIdx.x * 64 + wave * 16;
  const size_t bh = ((size_t)b * HH + h) * LL;
  const bf16* qp = Qb + bh * DD;
  const bf16* kp = Kb + bh * DD;
  const bf16* vp = Vb + bh * DD;

  auto stage_k = [&](int buf, int m0) {
#pragma unroll
    for (int i = 0; i < 2; ++i) {
      int idx = threadIdx.x + i * 128;
      int m = idx >> 3, c8 = (idx & 7) * 8;
      async_copy_b128(&k_lds[buf][m * 64 + c8],
                      &kp[(size_t)(m0 + m) * DD + c8]);
    }
  };
  bf16x8 vreg[2];
  auto load_v = [&](int m0) {
#pragma unroll
    for (int i = 0; i < 2; ++i) {
      int idx = threadIdx.x + i * 128;
      int m = idx >> 3, d8 = (idx & 7) * 8;
      vreg[i] = *reinterpret_cast<const bf16x8*>(&vp[(size_t)(m0 + m) * DD + d8]);
    }
  };
  auto store_vT = [&](int buf) {
#pragma unroll
    for (int i = 0; i < 2; ++i) {
      int idx = threadIdx.x + i * 128;
      int m = idx >> 3, d8 = (idx & 7) * 8;
#pragma unroll
      for (int j = 0; j < 8; ++j) vT_lds[buf][(d8 + j) * 32 + m] = vreg[i][j];
    }
  };

  // prologue: tile 0
  stage_k(0, 0);
  load_v(0);
  store_vT(0);

  // Q fragments for d=[0,32) and [32,64): live across the whole key loop.
  bf16x16 qf0 = load_afrag(qp, DD, l0 + ln, 0, half);
  bf16x16 qf1 = load_afrag(qp, DD, l0 + ln, 32, half);

  floatx8 o[4] = {};
  float rmax[8], rsum[8];
#pragma unroll
  for (int r = 0; r < 8; ++r) { rmax[r] = -1e30f; rsum[r] = 0.f; }

  for (int it = 0; it < LL / 32; ++it) {
    const int m0 = it * 32;
    const int cur = it & 1;
    wait_async0();        // this wave's K copies for `cur` complete
    __syncthreads();      // all waves' copies + V^T stores visible
    const bool nxt = (m0 + 32 < LL);
    if (nxt) {
      stage_k(cur ^ 1, m0 + 32);   // async, overlaps compute below
      load_v(m0 + 32);             // global loads in flight during compute
    }

    // S = Q K^T : two 16-key tiles, K-dim = D = 64 (2 WMMA each)
    floatx8 s[2];
#pragma unroll
    for (int j = 0; j < 2; ++j) {
      floatx8 sj = {};
      bf16x16 kb0 = load_bfrag(&k_lds[cur][(j * 16 + ln) * 64 + half * 16]);
      sj = wmma_bf16(qf0, kb0, sj);
      bf16x16 kb1 = load_bfrag(&k_lds[cur][(j * 16 + ln) * 64 + 32 + half * 16]);
      sj = wmma_bf16(qf1, kb1, sj);
      s[j] = sj;
    }

    // bias + online softmax (per-row stats across the 16-lane half-wave)
#pragma unroll
    for (int r = 0; r < 8; ++r) {
      int lg = l0 + r + 8 * half;
      float s0 = s[0][r] + attn_bias[(size_t)lg * LL + m0 + ln];
      float s1 = s[1][r] + attn_bias[(size_t)lg * LL + m0 + 16 + ln];
      float mx = redmax16(fmaxf(s0, s1));
      float mnew = fmaxf(rmax[r], mx);
      float corr = __expf(rmax[r] - mnew);
      float p0 = __expf(s0 - mnew);
      float p1 = __expf(s1 - mnew);
      rsum[r] = rsum[r] * corr + redsum16(p0 + p1);
      rmax[r] = mnew;
#pragma unroll
      for (int t = 0; t < 4; ++t) o[t][r] *= corr;
      int prow = r + 8 * half;
      p_lds[wave][prow * 32 + ln] = (bf16)p0;
      p_lds[wave][prow * 32 + 16 + ln] = (bf16)p1;
    }
    asm volatile("s_wait_dscnt 0x0" ::: "memory");  // P stores visible in-wave

    // P as A-fragment (K-dim = 32 keys)
    const bf16* pp = &p_lds[wave][0];
    bf16x8 plo = *reinterpret_cast<const bf16x8*>(&pp[ln * 32 + half * 16]);
    bf16x8 phi = *reinterpret_cast<const bf16x8*>(&pp[ln * 32 + half * 16 + 8]);
    bf16x16 pf = join8(plo, phi);

    // O += P V
#pragma unroll
    for (int t = 0; t < 4; ++t) {
      bf16x16 vb = load_bfrag(&vT_lds[cur][(t * 16 + ln) * 32 + half * 16]);
      o[t] = wmma_bf16(pf, vb, o[t]);
    }

    if (nxt) store_vT(cur ^ 1);    // transpose-store next V tile (late wait)
  }

  // epilogue: normalize by softmax denom, store O as [B, L, H, D] (= [B,L,C])
#pragma unroll
  for (int r = 0; r < 8; ++r) {
    float inv = 1.f / rsum[r];
    int lg = l0 + r + 8 * half;
    size_t base = (((size_t)b * LL + lg) * HH + h) * DD;
#pragma unroll
    for (int t = 0; t < 4; ++t) Ob[base + t * 16 + ln] = (bf16)(o[t][r] * inv);
  }
}

// ---------------------------------------------------------------------------
// Output projection: out = O @ w_proj^T + b_proj  (fp32 out)
// ---------------------------------------------------------------------------
__global__ __launch_bounds__(128) void proj_gemm_kernel(
    const bf16* __restrict__ ob, const bf16* __restrict__ wb,
    const float* __restrict__ b_proj, float* __restrict__ out) {
  __shared__ bf16 w_lds[2][64 * 32];
  const int lane = threadIdx.x & 31, wave = threadIdx.x >> 5;
  const int ln = lane & 15, half = lane >> 4;
  const int row0 = blockIdx.x * 64 + wave * 16;
  const int n0 = blockIdx.y * 64;

  auto stage_w = [&](int buf, int kk2) {
#pragma unroll
    for (int i = 0; i < 2; ++i) {
      int idx = threadIdx.x + i * 128;
      int n = idx >> 2, c8 = (idx & 3) * 8;
      async_copy_b128(&w_lds[buf][n * 32 + c8],
                      &wb[(size_t)(n0 + n) * CC + kk2 + c8]);
    }
  };

  stage_w(0, 0);
  bf16x16 afrag = load_afrag(ob, CC, row0 + ln, 0, half);

  floatx8 acc[4] = {};
  for (int kk = 0; kk < CC; kk += 32) {
    const int cur = (kk >> 5) & 1;
    wait_async0();
    __syncthreads();
    bf16x16 afrag_next = afrag;
    if (kk + 32 < CC) {
      stage_w(cur ^ 1, kk + 32);
      afrag_next = load_afrag(ob, CC, row0 + ln, kk + 32, half);
    }
#pragma unroll
    for (int t = 0; t < 4; ++t) {
      bf16x16 bfrag = load_bfrag(&w_lds[cur][(t * 16 + ln) * 32 + half * 16]);
      acc[t] = wmma_bf16(afrag, bfrag, acc[t]);
    }
    afrag = afrag_next;
  }
#pragma unroll
  for (int r = 0; r < 8; ++r) {
    int rg = row0 + r + 8 * half;
#pragma unroll
    for (int t = 0; t < 4; ++t) {
      int n = n0 + t * 16 + ln;
      out[(size_t)rg * CC + n] = acc[t][r] + b_proj[n];
    }
  }
}

// ---------------------------------------------------------------------------
// Launcher. Workspace layout (bf16 buffers), ~50.3 MB total:
//   xb[B*L*C] | wqkvb[3C*C] | wprojb[C*C] | Q | K | V | O  (each B*L*C)
// ---------------------------------------------------------------------------
extern "C" void kernel_launch(void* const* d_in, const int* in_sizes, int n_in,
                              void* d_out, int out_size, void* d_ws, size_t ws_size,
                              hipStream_t stream) {
  const float* x         = (const float*)d_in[0];
  const float* attn_bias = (const float*)d_in[1];
  const float* w_qkv     = (const float*)d_in[2];
  const float* q_bias    = (const float*)d_in[3];
  const float* v_bias    = (const float*)d_in[4];
  const float* scale_mul = (const float*)d_in[5];
  const float* w_proj    = (const float*)d_in[6];
  const float* b_proj    = (const float*)d_in[7];
  float* out = (float*)d_out;

  const size_t NXC = (size_t)BB * LL * CC;        // 4,194,304
  const size_t NWQ = (size_t)3 * CC * CC;         // 3,145,728
  const size_t NWP = (size_t)CC * CC;             // 1,048,576

  char* ws = (char*)d_ws;
  size_t off = 0;
  auto alloc = [&](size_t elems) { bf16* p = (bf16*)(ws + off);
                                   off += ((elems * sizeof(bf16) + 255) & ~size_t(255));
                                   return p; };
  bf16* xb     = alloc(NXC);
  bf16* wqkvb  = alloc(NWQ);
  bf16* wprojb = alloc(NWP);
  bf16* Qb     = alloc(NXC);
  bf16* Kb     = alloc(NXC);
  bf16* Vb     = alloc(NXC);
  bf16* Ob     = alloc(NXC);

  // 1) fp32 -> bf16
  cvt_bf16_kernel<<<(int)(NXC / 4 + 255) / 256, 256, 0, stream>>>(x, xb, (int)(NXC / 4));
  cvt_bf16_kernel<<<(int)(NWQ / 4 + 255) / 256, 256, 0, stream>>>(w_qkv, wqkvb, (int)(NWQ / 4));
  cvt_bf16_kernel<<<(int)(NWP / 4 + 255) / 256, 256, 0, stream>>>(w_proj, wprojb, (int)(NWP / 4));

  // 2) QKV GEMM + fused bias / L2-norm / scale
  qkv_gemm_kernel<<<dim3((BB * LL) / 64, (3 * CC) / 64), 128, 0, stream>>>(
      xb, wqkvb, q_bias, v_bias, scale_mul, Qb, Kb, Vb);

  // 3) flash attention
  attn_kernel<<<dim3(LL / 64, HH, BB), 128, 0, stream>>>(Qb, Kb, Vb, attn_bias, Ob);

  // 4) output projection
  proj_gemm_kernel<<<dim3((BB * LL) / 64, CC / 64), 128, 0, stream>>>(
      Ob, wprojb, b_proj, out);
}
